// ConditionalLayer_16475494548255
// MI455X (gfx1250) — matrix operand order repaired and checked by reading, not compiled, not run on hardware
//
#include <hip/hip_runtime.h>

// ---------------------------------------------------------------------------
// ConditionalLayer for MI455X (gfx1250, wave32, WMMA + async-to-LDS).
//   out[b] = relu(x[b]W1[cb]+b1[cb])W2[cb]+b2[cb] + const_all - cterm[cb]
//   cterm[c] = relu(b1[c])W2[c] + b2[c],  const_all = sum_c cterm[c]
// Grouped GEMM over condition buckets. fp32 emulated via bf16 hi/lo split
// (3x v_wmma_f32_16x16x32_bf16 per fragment pair, ~2^-16 rel err).
// All fp32->bf16 splitting hoisted into one-time pre-passes; GEMM hot loop is
// pure global_load_async_to_lds_b128 (ASYNCcnt, LDS double buffer) + WMMA.
// ---------------------------------------------------------------------------

#define B_ROWS 4096
#define DIM    1024
#define NCOND  8
#define BM     128
#define BN     64
#define BK     32

typedef __attribute__((ext_vector_type(16))) __bf16 v16bf;
typedef __attribute__((ext_vector_type(8)))  __bf16 v8bf;
typedef __attribute__((ext_vector_type(4)))  __bf16 v4bf;
typedef __attribute__((ext_vector_type(8)))  float  v8f;
typedef __attribute__((ext_vector_type(4)))  float  v4f;

// ---- CDNA5 async copy: LDS[lds_off] = MEM[saddr + goff], 16B per lane ------
__device__ __forceinline__ void async_b128(unsigned lds_off, unsigned goff,
                                           const void* base) {
  asm volatile("global_load_async_to_lds_b128 %0, %1, %2"
               :: "v"(lds_off), "v"(goff), "s"(base) : "memory");
}
__device__ __forceinline__ void wait_async6() {
  asm volatile("s_wait_asynccnt 6" ::: "memory");
}
__device__ __forceinline__ void wait_async0() {
  asm volatile("s_wait_asynccnt 0" ::: "memory");
}
__device__ __forceinline__ unsigned lds_off_of(const void* p) {
  return (unsigned)(uintptr_t)p;  // generic LDS addr: low 32 bits = LDS offset
}

// ---------------------------------------------------------------------------
// Kernel 1: bucket rows by condition, build group offsets and M-tile map.
// ---------------------------------------------------------------------------
__global__ void bucket_kernel(const int* __restrict__ cond,
                              int* __restrict__ row_idx,
                              int* __restrict__ grp_off,
                              int* __restrict__ tile_c,
                              int* __restrict__ tile_m,
                              int* __restrict__ n_tiles) {
  __shared__ int cnt[NCOND], cur[NCOND], off[NCOND + 1];
  const int tid = threadIdx.x;
  if (tid < NCOND) cnt[tid] = 0;
  __syncthreads();
  for (int b = tid; b < B_ROWS; b += 256) atomicAdd(&cnt[cond[b]], 1);
  __syncthreads();
  if (tid == 0) {
    off[0] = 0;
    int t = 0;
    for (int c = 0; c < NCOND; ++c) {
      off[c + 1] = off[c] + cnt[c];
      cur[c] = off[c];
      grp_off[c] = off[c];
      for (int m = off[c]; m < off[c + 1]; m += BM) {
        tile_c[t] = c;
        tile_m[t] = m;
        ++t;
      }
    }
    grp_off[NCOND] = off[NCOND];
    *n_tiles = t;  // <= B/BM + NCOND = 40
  }
  __syncthreads();
  for (int b = tid; b < B_ROWS; b += 256) {
    int c = cond[b];
    int pos = atomicAdd(&cur[c], 1);
    row_idx[pos] = b;  // order within group irrelevant: per-row math is slot-independent
  }
}

// ---------------------------------------------------------------------------
// Kernel 2: cterm[c][e] = relu(b1[c]) . W2[c][:,e] + b2[c][e]
// Skips the 32MB W2 read when relu(b1[c]) == 0 (true for the given inputs).
// ---------------------------------------------------------------------------
__global__ void cterm_kernel(const float* __restrict__ b1,
                             const float* __restrict__ b2,
                             const float* __restrict__ W2,
                             float* __restrict__ cterm) {
  __shared__ float rb1[DIM];
  __shared__ int any_nz;
  const int c = blockIdx.x;
  const int tid = threadIdx.x;
  if (tid == 0) any_nz = 0;
  __syncthreads();
  int local = 0;
  for (int d = tid; d < DIM; d += 256) {
    float v = b1[c * DIM + d];
    v = v > 0.f ? v : 0.f;
    rb1[d] = v;
    if (v != 0.f) local = 1;
  }
  if (local) atomicOr(&any_nz, 1);
  __syncthreads();
  const int e = blockIdx.y * 256 + tid;
  float acc = b2[c * DIM + e];
  if (any_nz) {
    const float* w = W2 + (size_t)c * DIM * DIM + e;
    for (int d = 0; d < DIM; ++d) acc += rb1[d] * w[(size_t)d * DIM];
  }
  cterm[c * DIM + e] = acc;
}

__global__ void constall_kernel(const float* __restrict__ cterm,
                                float* __restrict__ const_all) {
  const int e = blockIdx.x * 256 + threadIdx.x;
  float s = 0.f;
  for (int c = 0; c < NCOND; ++c) s += cterm[c * DIM + e];
  const_all[e] = s;
}

// ---------------------------------------------------------------------------
// Kernel 3: split x into bf16 hi/lo planes; zero the 64B gather pads of the
// x and H planes (used by ragged-tile lanes so ASYNCcnt stays exact).
// ---------------------------------------------------------------------------
__global__ void split_x_kernel(const float* __restrict__ x,
                               __bf16* __restrict__ xhi, __bf16* __restrict__ xlo,
                               __bf16* __restrict__ hhi, __bf16* __restrict__ hlo) {
  const int i = (blockIdx.x * 256 + threadIdx.x) * 4;
  v4f v = *(const v4f*)(x + i);
  v4bf h, l;
#pragma unroll
  for (int j = 0; j < 4; ++j) {
    __bf16 hh = (__bf16)v[j];
    h[j] = hh;
    l[j] = (__bf16)(v[j] - (float)hh);
  }
  *(v4bf*)(xhi + i) = h;
  *(v4bf*)(xlo + i) = l;
  if (blockIdx.x == 0 && threadIdx.x < 32) {
    const int t = B_ROWS * DIM + threadIdx.x;
    xhi[t] = (__bf16)0.f;
    xlo[t] = (__bf16)0.f;
    hhi[t] = (__bf16)0.f;
    hlo[t] = (__bf16)0.f;
  }
}

// ---------------------------------------------------------------------------
// Kernel 4: split+transpose weights: W[c][k][n] fp32 -> Whi/Wlo[c][n][k] bf16.
// 64x64 tiles through LDS so both global read and write are coalesced.
// ---------------------------------------------------------------------------
__global__ void __launch_bounds__(256)
split_w_kernel(const float* __restrict__ W,
               __bf16* __restrict__ Whi, __bf16* __restrict__ Wlo) {
  __shared__ __bf16 shi[64][72];  // pad to 72 (144B rows, 16B-aligned chunks)
  __shared__ __bf16 slo[64][72];
  const int c = blockIdx.z;
  const int k0 = blockIdx.y * 64;
  const int n0 = blockIdx.x * 64;
  const int tid = threadIdx.x;
  const int kr = tid >> 4;          // + pass*16
  const int nc = (tid & 15) * 4;
#pragma unroll
  for (int pass = 0; pass < 4; ++pass) {
    const int kk = kr + pass * 16;
    v4f v = *(const v4f*)(W + ((size_t)(c * DIM + k0 + kk)) * DIM + n0 + nc);
#pragma unroll
    for (int j = 0; j < 4; ++j) {
      __bf16 h = (__bf16)v[j];
      shi[nc + j][kk] = h;
      slo[nc + j][kk] = (__bf16)(v[j] - (float)h);
    }
  }
  __syncthreads();
#pragma unroll
  for (int pass = 0; pass < 2; ++pass) {
    const int idx = pass * 256 + tid;
    const int r = idx >> 3, seg = (idx & 7) * 8;
    const size_t o = ((size_t)(c * DIM + n0 + r)) * DIM + k0 + seg;
    *(v8bf*)(Whi + o) = *(const v8bf*)&shi[r][seg];
    *(v8bf*)(Wlo + o) = *(const v8bf*)&slo[r][seg];
  }
}

// ---------------------------------------------------------------------------
// Grouped GEMM layer. Block = 256 threads = 8 waves (4x2 wave grid), each
// wave a 32x32 region (2x2 C fragments) -> 12 WMMAs per K-step.
// Staging: 6 async b128 copies/wave/K-step into LDS double buffer; the next
// slab flies under the current slab's WMMAs (in-order ASYNCcnt <= 6 wait).
// ---------------------------------------------------------------------------
__device__ __forceinline__ v16bf cat16(v8bf a, v8bf b) {
  return __builtin_shufflevector(a, b, 0, 1, 2, 3, 4, 5, 6, 7,
                                 8, 9, 10, 11, 12, 13, 14, 15);
}

template <bool LAYER2>
__global__ void __launch_bounds__(256)
fc_gemm_kernel(const __bf16* __restrict__ A_hi, const __bf16* __restrict__ A_lo,
               const __bf16* __restrict__ W_hi, const __bf16* __restrict__ W_lo,
               const float* __restrict__ bias,
               const float* __restrict__ const_all, // L2 only
               const float* __restrict__ cterm,     // L2 only
               const int* __restrict__ row_idx,
               const int* __restrict__ grp_off,
               const int* __restrict__ tile_c,
               const int* __restrict__ tile_m,
               const int* __restrict__ n_tiles_p,
               __bf16* __restrict__ OutHi,          // L1: H hi plane
               __bf16* __restrict__ OutLo,          // L1: H lo plane
               float* __restrict__ OutF) {          // L2: final output
  const int ty = blockIdx.y;
  if (ty >= *n_tiles_p) return;  // uniform early-out, before any barrier

  const int c      = tile_c[ty];
  const int m_base = tile_m[ty];
  const int m_end  = grp_off[c + 1];
  const int n0     = blockIdx.x * BN;

  __shared__ __bf16 lA_hi[2][BM][BK];
  __shared__ __bf16 lA_lo[2][BM][BK];
  __shared__ __bf16 lW_hi[2][BN][BK];
  __shared__ __bf16 lW_lo[2][BN][BK];

  const int tid    = threadIdx.x;
  const int lane   = tid & 31;
  const int wave   = tid >> 5;
  const int wave_m = wave & 3;   // M rows [wave_m*32, +32)
  const int wave_n = wave >> 2;  // N cols [wave_n*32, +32)

  // ---- async copy descriptors (byte offsets; advance per K-slab) ----
  // A: 128 rows x 64B per plane = 512 x 16B chunks -> 2 rounds of 256 lanes.
  const unsigned ZOFF = (unsigned)(B_ROWS * DIM * 2);  // zero-pad gather target
  unsigned a_goff[2], a_adv[2], a_lds[2];
#pragma unroll
  for (int r2 = 0; r2 < 2; ++r2) {
    const int idx = r2 * 256 + tid;
    const int row = idx >> 2;          // 0..127
    const int seg = idx & 3;           // 16B segment within the 64B k-slab
    const int p = m_base + row;
    const bool ok = p < m_end;
    unsigned grow = 0;
    if (ok) grow = LAYER2 ? (unsigned)p : (unsigned)row_idx[p];
    a_goff[r2] = ok ? (grow * (unsigned)DIM + seg * 8u) * 2u : ZOFF + seg * 16u;
    a_adv[r2]  = ok ? (unsigned)(BK * 2) : 0u;   // pads don't advance with k
    a_lds[r2]  = (unsigned)(row * (BK * 2) + seg * 16);
  }
  // W: 64 rows x 64B per plane = 256 x 16B chunks -> 1 round.
  const int wrow = tid >> 2, wseg = tid & 3;
  unsigned w_goff = ((unsigned)(c * DIM + n0 + wrow) * (unsigned)DIM + wseg * 8u) * 2u;
  const unsigned w_lds = (unsigned)(wrow * (BK * 2) + wseg * 16);

  const unsigned baseAhi = lds_off_of(&lA_hi[0][0][0]);
  const unsigned baseAlo = lds_off_of(&lA_lo[0][0][0]);
  const unsigned baseWhi = lds_off_of(&lW_hi[0][0][0]);
  const unsigned baseWlo = lds_off_of(&lW_lo[0][0][0]);
  const unsigned szA = BM * BK * 2, szW = BN * BK * 2;

  auto issue_slab = [&](int buf) {  // exactly 6 async ops per wave
#pragma unroll
    for (int r2 = 0; r2 < 2; ++r2) {
      async_b128(baseAhi + buf * szA + a_lds[r2], a_goff[r2], A_hi);
      async_b128(baseAlo + buf * szA + a_lds[r2], a_goff[r2], A_lo);
    }
    async_b128(baseWhi + buf * szW + w_lds, w_goff, W_hi);
    async_b128(baseWlo + buf * szW + w_lds, w_goff, W_lo);
    a_goff[0] += a_adv[0];
    a_goff[1] += a_adv[1];
    w_goff += BK * 2;
  };

  // Fragment index math (ISA 7.12.2 layouts, wave32):
  const int fm   = lane & 15;         // A row within 16
  const int ksel = (lane >> 4) * 8;   // A: k in {ksel..ksel+7, ksel+16..ksel+23}
  const int kb2  = (lane >> 4) * 16;  // unused; B uses ksel (same k order as A)

  v8f acc[2][2] = {{v8f{}, v8f{}}, {v8f{}, v8f{}}};

  issue_slab(0);
  for (int k0 = 0; k0 < DIM; k0 += BK) {
    const int cur = (k0 >> 5) & 1;
    if (k0 + BK < DIM) {
      issue_slab(cur ^ 1);   // next slab flies under this slab's WMMAs
      wait_async6();         // in-order: all but the 6 newest are complete
    } else {
      wait_async0();
    }
    __syncthreads();         // every wave's slab-k copies have landed

    v16bf bh[2], bl[2];
#pragma unroll
    for (int in = 0; in < 2; ++in) {
      const int bn = wave_n * 32 + in * 16 + (lane & 15);
      bh[in] = cat16(*(const v8bf*)&lW_hi[cur][bn][ksel],
                     *(const v8bf*)&lW_hi[cur][bn][ksel + 16]);
      bl[in] = cat16(*(const v8bf*)&lW_lo[cur][bn][ksel],
                     *(const v8bf*)&lW_lo[cur][bn][ksel + 16]);
    }
#pragma unroll
    for (int im = 0; im < 2; ++im) {
      const int ar = wave_m * 32 + im * 16 + fm;
      const v16bf ah = cat16(*(const v8bf*)&lA_hi[cur][ar][ksel],
                             *(const v8bf*)&lA_hi[cur][ar][ksel + 16]);
      const v16bf al = cat16(*(const v8bf*)&lA_lo[cur][ar][ksel],
                             *(const v8bf*)&lA_lo[cur][ar][ksel + 16]);
#pragma unroll
      for (int in = 0; in < 2; ++in) {
        acc[im][in] = __builtin_amdgcn_wmma_f32_16x16x32_bf16(
            false, ah, false, bh[in], (short)0, acc[im][in], false, false);
        acc[im][in] = __builtin_amdgcn_wmma_f32_16x16x32_bf16(
            false, ah, false, bl[in], (short)0, acc[im][in], false, false);
        acc[im][in] = __builtin_amdgcn_wmma_f32_16x16x32_bf16(
            false, al, false, bh[in], (short)0, acc[im][in], false, false);
      }
    }
    __syncthreads();         // all reads done before buf is overwritten at k+2
  }

  // ---- epilogue ----
#pragma unroll
  for (int in = 0; in < 2; ++in) {
    const int col = n0 + wave_n * 32 + in * 16 + (lane & 15);
    float badd = bias[c * DIM + col];
    if (LAYER2) badd += const_all[col] - cterm[c * DIM + col];
#pragma unroll
    for (int im = 0; im < 2; ++im) {
#pragma unroll
      for (int r = 0; r < 8; ++r) {
        const int mrow = wave_m * 32 + im * 16 + r + 8 * (lane >> 4);
        const int p = m_base + mrow;
        if (p < m_end) {
          float v = acc[im][in][r] + badd;
          if (!LAYER2) {
            v = v > 0.f ? v : 0.f;  // relu, then store H as split bf16 planes
            __bf16 h = (__bf16)v;
            OutHi[(size_t)p * DIM + col] = h;
            OutLo[(size_t)p * DIM + col] = (__bf16)(v - (float)h);
          } else {
            OutF[(size_t)row_idx[p] * DIM + col] = v;
          }
        }
      }
    }
  }
}

// ---------------------------------------------------------------------------
// Host launcher.
// ---------------------------------------------------------------------------
extern "C" void kernel_launch(void* const* d_in, const int* in_sizes, int n_in,
                              void* d_out, int out_size, void* d_ws, size_t ws_size,
                              hipStream_t stream) {
  (void)in_sizes; (void)n_in; (void)out_size; (void)ws_size;
  const float* x    = (const float*)d_in[0];
  const int*   cond = (const int*)d_in[1];
  const float* W1   = (const float*)d_in[2];
  const float* b1   = (const float*)d_in[3];
  const float* W2   = (const float*)d_in[4];
  const float* b2   = (const float*)d_in[5];
  float* out = (float*)d_out;

  char* ws = (char*)d_ws;
  size_t off = 0;
  auto carve = [&](size_t bytes) -> char* {
    char* p = ws + off;
    off = (off + bytes + 255) & ~(size_t)255;
    return p;
  };
  const size_t planeB = ((size_t)B_ROWS * DIM + 32) * sizeof(__bf16);  // +64B pad
  int*    row_idx   = (int*)carve(B_ROWS * sizeof(int));
  int*    grp_off   = (int*)carve((NCOND + 1) * sizeof(int));
  int*    tile_c    = (int*)carve(80 * sizeof(int));
  int*    tile_m    = (int*)carve(80 * sizeof(int));
  int*    n_tiles   = (int*)carve(sizeof(int));
  float*  cterm     = (float*)carve((size_t)NCOND * DIM * sizeof(float));
  float*  const_all = (float*)carve((size_t)DIM * sizeof(float));
  __bf16* x_hi      = (__bf16*)carve(planeB);
  __bf16* x_lo      = (__bf16*)carve(planeB);
  __bf16* H_hi      = (__bf16*)carve(planeB);
  __bf16* H_lo      = (__bf16*)carve(planeB);
  __bf16* Wt_hi     = (__bf16*)carve((size_t)NCOND * DIM * DIM * sizeof(__bf16));
  __bf16* Wt_lo     = (__bf16*)carve((size_t)NCOND * DIM * DIM * sizeof(__bf16));

  bucket_kernel<<<1, 256, 0, stream>>>(cond, row_idx, grp_off, tile_c, tile_m, n_tiles);
  cterm_kernel<<<dim3(NCOND, DIM / 256), 256, 0, stream>>>(b1, b2, W2, cterm);
  constall_kernel<<<DIM / 256, 256, 0, stream>>>(cterm, const_all);
  split_x_kernel<<<(B_ROWS * DIM) / 1024, 256, 0, stream>>>(x, x_hi, x_lo, H_hi, H_lo);

  // Layer 1 (Wt buffer holds split W1, then reused for W2 — stream-ordered).
  split_w_kernel<<<dim3(DIM / 64, DIM / 64, NCOND), 256, 0, stream>>>(W1, Wt_hi, Wt_lo);
  fc_gemm_kernel<false><<<dim3(DIM / BN, 40), 256, 0, stream>>>(
      x_hi, x_lo, Wt_hi, Wt_lo, b1, nullptr, nullptr,
      row_idx, grp_off, tile_c, tile_m, n_tiles, H_hi, H_lo, nullptr);
  split_w_kernel<<<dim3(DIM / 64, DIM / 64, NCOND), 256, 0, stream>>>(W2, Wt_hi, Wt_lo);
  fc_gemm_kernel<true><<<dim3(DIM / BN, 40), 256, 0, stream>>>(
      H_hi, H_lo, Wt_hi, Wt_lo, b2, const_all, cterm,
      row_idx, grp_off, tile_c, tile_m, n_tiles, nullptr, nullptr, out);
}
